// MambaBlock_75058848465499
// MI455X (gfx1250) — compile-verified, matrix operand order checked by reading, not run on hardware
//
#include <hip/hip_runtime.h>
#include <hip/hip_bf16.h>

// ---------------------------------------------------------------------------
// Mamba block for MI455X (gfx1250, wave32, WMMA).
// GEMMs in bf16 via v_wmma_f32_16x16x32_bf16 with register double-buffering
// (loads for k+32 issued before WMMAs for k) and 32x64 output per wave for
// higher compute/load ratio. Selective scan keeps h-states in VGPRs with
// A/B/C broadcast from LDS and exp on the TRANS pipe.
// ---------------------------------------------------------------------------

typedef __attribute__((ext_vector_type(16))) __bf16 v16bf;
typedef __attribute__((ext_vector_type(8)))  float  v8f;

constexpr int kBatch  = 4;
constexpr int kSeq    = 4096;
constexpr int kDModel = 1024;
constexpr int kDInner = 2048;
constexpr int kDState = 64;
constexpr int kRows   = kBatch * kSeq;        // 16384

// ---------------------------------------------------------------------------
// Conversion / layout kernels
// ---------------------------------------------------------------------------
__global__ __launch_bounds__(256) void f32_to_bf16_kernel(
    const float* __restrict__ in, __hip_bfloat16* __restrict__ out, size_t n) {
  size_t i = (size_t)blockIdx.x * 256 + threadIdx.x;
  if (i < n) out[i] = __float2bfloat16(in[i]);
}

// in: R x C (f32, row-major)  ->  out: C x R (bf16, row-major)  [weight W^T]
__global__ __launch_bounds__(256) void transpose_to_bf16_kernel(
    const float* __restrict__ in, __hip_bfloat16* __restrict__ out, int R, int C) {
  size_t i = (size_t)blockIdx.x * 256 + threadIdx.x;
  size_t total = (size_t)R * C;
  if (i >= total) return;
  int c = (int)(i / R);
  int r = (int)(i % R);
  out[i] = __float2bfloat16(in[(size_t)r * C + c]);
}

// Pack [sB_w | sC_w] (each D_INNER x 64, f32) into transposed bf16 (128 x D_INNER)
__global__ __launch_bounds__(256) void pack_bc_kernel(
    const float* __restrict__ sB, const float* __restrict__ sC,
    __hip_bfloat16* __restrict__ out) {
  size_t i = (size_t)blockIdx.x * 256 + threadIdx.x;       // 128 * 2048
  if (i >= (size_t)128 * kDInner) return;
  int n = (int)(i / kDInner);
  int k = (int)(i % kDInner);
  float v = (n < kDState) ? sB[(size_t)k * kDState + n]
                          : sC[(size_t)k * kDState + (n - kDState)];
  out[i] = __float2bfloat16(v);
}

// ---------------------------------------------------------------------------
// bf16 WMMA GEMM:  C[M,N] (f32) = A[M,K] (bf16) * Bt[N,K]^T (bf16)  + epilogue
// One wave -> 32x64 strip: 2 A-frags x 4 B-frags -> 8 WMMAs / 6 fragment
// loads per 32-wide K step. Fragments are double-buffered in registers so the
// s_wait for a step's loads is issued a full iteration after the loads.
// 8 waves/block -> 256x64 block tile.
// EPI: 0 = none, 1 = softplus(x + bias), 2 = x + bias
// ---------------------------------------------------------------------------
#define WMMA_BF16(A_, B_, C_)                                            \
  __builtin_amdgcn_wmma_f32_16x16x32_bf16(false, (A_), false, (B_),      \
                                          (short)0, (C_), false, false)

template <int EPI>
__global__ __launch_bounds__(256) void gemm_bf16_kernel(
    const __hip_bfloat16* __restrict__ A,   // M x K
    const __hip_bfloat16* __restrict__ Bt,  // N x K (pre-transposed weights)
    float* __restrict__ C,                  // M x N
    const float* __restrict__ bias,         // N (or nullptr)
    int M, int N, int K) {
  const int lane = threadIdx.x & 31;
  const int wave = threadIdx.x >> 5;
  const int m0 = blockIdx.y * 256 + wave * 32;
  const int n0 = blockIdx.x * 64;
  const int mr = lane & 15;               // A row / B column within fragment
  const int kh = (lane >> 4) * 16;        // K half selected by lane group

  const __hip_bfloat16* ap0 = A  + (size_t)(m0 + mr) * K + kh;
  const __hip_bfloat16* ap1 = A  + (size_t)(m0 + 16 + mr) * K + kh;
  const __hip_bfloat16* bp  = Bt + (size_t)(n0 + mr) * K + kh;
  const size_t bs = (size_t)16 * K;

  v8f acc00 = v8f(0.f), acc01 = v8f(0.f), acc02 = v8f(0.f), acc03 = v8f(0.f);
  v8f acc10 = v8f(0.f), acc11 = v8f(0.f), acc12 = v8f(0.f), acc13 = v8f(0.f);

  // Prologue: fragments for k = 0
  v16bf a0 = *(const v16bf*)(ap0);
  v16bf a1 = *(const v16bf*)(ap1);
  v16bf b0 = *(const v16bf*)(bp);
  v16bf b1 = *(const v16bf*)(bp + bs);
  v16bf b2 = *(const v16bf*)(bp + 2 * bs);
  v16bf b3 = *(const v16bf*)(bp + 3 * bs);

  int k = 0;
  for (; k + 32 < K; k += 32) {
    __builtin_prefetch(ap0 + k + 256, 0, 3);   // keep-near prefetch
    // Issue next step's loads first (distinct registers -> latency hidden
    // behind this step's 8 WMMAs).
    v16bf a0n = *(const v16bf*)(ap0 + k + 32);
    v16bf a1n = *(const v16bf*)(ap1 + k + 32);
    v16bf b0n = *(const v16bf*)(bp + k + 32);
    v16bf b1n = *(const v16bf*)(bp + bs + k + 32);
    v16bf b2n = *(const v16bf*)(bp + 2 * bs + k + 32);
    v16bf b3n = *(const v16bf*)(bp + 3 * bs + k + 32);

    acc00 = WMMA_BF16(a0, b0, acc00);
    acc10 = WMMA_BF16(a1, b0, acc10);
    acc01 = WMMA_BF16(a0, b1, acc01);
    acc11 = WMMA_BF16(a1, b1, acc11);
    acc02 = WMMA_BF16(a0, b2, acc02);
    acc12 = WMMA_BF16(a1, b2, acc12);
    acc03 = WMMA_BF16(a0, b3, acc03);
    acc13 = WMMA_BF16(a1, b3, acc13);

    a0 = a0n; a1 = a1n;
    b0 = b0n; b1 = b1n; b2 = b2n; b3 = b3n;
  }
  // Epilogue K-step
  acc00 = WMMA_BF16(a0, b0, acc00);
  acc10 = WMMA_BF16(a1, b0, acc10);
  acc01 = WMMA_BF16(a0, b1, acc01);
  acc11 = WMMA_BF16(a1, b1, acc11);
  acc02 = WMMA_BF16(a0, b2, acc02);
  acc12 = WMMA_BF16(a1, b2, acc12);
  acc03 = WMMA_BF16(a0, b3, acc03);
  acc13 = WMMA_BF16(a1, b3, acc13);

  // D layout: lanes 0-15 hold M = base + r, lanes 16-31 hold M = base + 8 + r;
  // N = lane & 15.
  const int nc = lane & 15;
  const int mhalf = (lane >> 4) * 8;
#pragma unroll
  for (int i = 0; i < 2; ++i) {
    const int mbase = m0 + 16 * i + mhalf;
#pragma unroll
    for (int j = 0; j < 4; ++j) {
      v8f acc;
      if (i == 0) acc = (j == 0) ? acc00 : (j == 1) ? acc01 : (j == 2) ? acc02 : acc03;
      else        acc = (j == 0) ? acc10 : (j == 1) ? acc11 : (j == 2) ? acc12 : acc13;
      const int n = n0 + j * 16 + nc;
      float bv = (EPI != 0) ? bias[n] : 0.f;
#pragma unroll
      for (int r = 0; r < 8; ++r) {
        float v = acc[r];
        if (EPI == 1) {                  // softplus(v + bias), numerically stable
          v += bv;
          v = (v > 20.f) ? v : log1pf(__expf(v));
        } else if (EPI == 2) {
          v += bv;
        }
        C[(size_t)(mbase + r) * N + n] = v;
      }
    }
  }
}

// ---------------------------------------------------------------------------
// Causal depthwise conv1d (k=4) + bias + SiLU.  Reads x_ssm half of x_proj.
// Emits xc as f32 (for scan) and bf16 (for the delta and [B | C] GEMMs).
// ---------------------------------------------------------------------------
__global__ __launch_bounds__(256) void conv_silu_kernel(
    const float* __restrict__ xproj,       // (B*L) x 4096; x_ssm = cols [0,2048)
    const float* __restrict__ cw,          // 2048 x 4
    const float* __restrict__ cb,          // 2048
    float* __restrict__ xc,                // (B*L) x 2048
    __hip_bfloat16* __restrict__ xcb) {
  const int d = blockIdx.x * 256 + threadIdx.x;
  const int t = blockIdx.y;
  const int b = blockIdx.z;
  float acc = cb[d];
#pragma unroll
  for (int j = 0; j < 4; ++j) {
    int ts = t - 3 + j;
    if (ts >= 0)
      acc += cw[d * 4 + j] * xproj[((size_t)b * kSeq + ts) * (2 * kDInner) + d];
  }
  float s = acc / (1.f + __expf(-acc));    // SiLU
  size_t idx = ((size_t)b * kSeq + t) * kDInner + d;
  xc[idx]  = s;
  xcb[idx] = __float2bfloat16(s);
}

// ---------------------------------------------------------------------------
// Selective scan. One thread per (b, d) channel; h[64] lives in VGPRs.
// A[64], B(t)[64], C(t)[64] broadcast from LDS (uniform address -> bank bcast).
// Fused epilogue: y = (scan + xc*D) * silu(gate), emitted as bf16.
// ---------------------------------------------------------------------------
__global__ __launch_bounds__(256) void scan_kernel(
    const float* __restrict__ delta,       // (B*L) x 2048
    const float* __restrict__ xc,          // (B*L) x 2048
    const float* __restrict__ BC,          // (B*L) x 128   [B | C]
    const float* __restrict__ xproj,       // (B*L) x 4096  (gate = cols [2048,4096))
    const float* __restrict__ A_log,       // 64
    const float* __restrict__ D_param,     // 2048
    __hip_bfloat16* __restrict__ yb) {     // (B*L) x 2048
  __shared__ float sA[kDState];
  __shared__ float sBC[2 * kDState];
  const int tid = threadIdx.x;
  const int b = blockIdx.x;
  const int d = blockIdx.y * 256 + tid;

  if (tid < kDState) sA[tid] = -__expf(A_log[tid]);

  float h[kDState];
#pragma unroll
  for (int n = 0; n < kDState; ++n) h[n] = 0.f;
  const float Dd = D_param[d];

  for (int t = 0; t < kSeq; ++t) {
    const size_t bt = (size_t)b * kSeq + t;
    __syncthreads();
    if (tid < 128) {
      sBC[tid] = BC[bt * 128 + tid];
      if (t + 8 < kSeq) __builtin_prefetch(&BC[(bt + 8) * 128 + tid], 0, 3);
    }
    __syncthreads();

    const float dt = delta[bt * kDInner + d];
    const float xt = xc[bt * kDInner + d];
    const float dx = dt * xt;
    float y = 0.f;
#pragma unroll
    for (int n = 0; n < kDState; ++n) {
      float ab = __expf(dt * sA[n]);                 // TRANS pipe, co-executes
      h[n] = fmaf(ab, h[n], dx * sBC[n]);
      y = fmaf(h[n], sBC[kDState + n], y);
    }

    float g = xproj[bt * (2 * kDInner) + kDInner + d];
    float gate = g / (1.f + __expf(-g));             // silu(gate)
    float out = (y + xt * Dd) * gate;
    yb[bt * kDInner + d] = __float2bfloat16(out);
  }
}

// ---------------------------------------------------------------------------
// Host-side orchestration
// ---------------------------------------------------------------------------
extern "C" void kernel_launch(void* const* d_in, const int* in_sizes, int n_in,
                              void* d_out, int out_size, void* d_ws, size_t ws_size,
                              hipStream_t stream) {
  (void)in_sizes; (void)n_in; (void)out_size; (void)ws_size;

  const float* x        = (const float*)d_in[0];   // (B,L,1024)
  const float* in_w     = (const float*)d_in[1];   // (1024, 4096)
  const float* conv_w   = (const float*)d_in[2];   // (2048, 1, 4)
  const float* conv_b   = (const float*)d_in[3];   // (2048,)
  const float* sB_w     = (const float*)d_in[4];   // (2048, 64)
  const float* sC_w     = (const float*)d_in[5];   // (2048, 64)
  const float* sdelta_w = (const float*)d_in[6];   // (2048, 2048)
  const float* sdelta_b = (const float*)d_in[7];   // (2048,)
  const float* A_log    = (const float*)d_in[8];   // (64,)
  const float* D_param  = (const float*)d_in[9];   // (2048,)
  const float* out_w    = (const float*)d_in[10];  // (2048, 1024)
  const float* out_b    = (const float*)d_in[11];  // (1024,)
  float* out = (float*)d_out;                      // (B,L,1024)

  char* ws = (char*)d_ws;
  size_t off = 0;
  auto take = [&](size_t bytes) -> char* {
    off = (off + 255) & ~(size_t)255;
    char* p = ws + off;
    off += bytes;
    return p;
  };

  __hip_bfloat16* xb    = (__hip_bfloat16*)take((size_t)kRows * kDModel * 2);
  __hip_bfloat16* wInT  = (__hip_bfloat16*)take((size_t)(2 * kDInner) * kDModel * 2);
  __hip_bfloat16* wDT   = (__hip_bfloat16*)take((size_t)kDInner * kDInner * 2);
  __hip_bfloat16* wBCT  = (__hip_bfloat16*)take((size_t)128 * kDInner * 2);
  __hip_bfloat16* wOutT = (__hip_bfloat16*)take((size_t)kDModel * kDInner * 2);
  float*          xproj = (float*)take((size_t)kRows * (2 * kDInner) * 4);
  float*          xc    = (float*)take((size_t)kRows * kDInner * 4);
  __hip_bfloat16* xcb   = (__hip_bfloat16*)take((size_t)kRows * kDInner * 2);
  float*          delta = (float*)take((size_t)kRows * kDInner * 4);
  float*          BC    = (float*)take((size_t)kRows * 128 * 4);
  __hip_bfloat16* yb    = (__hip_bfloat16*)take((size_t)kRows * kDInner * 2);

  // 1) Precision conversion / weight transposition (one-time layout pass)
  {
    size_t n = (size_t)kRows * kDModel;
    f32_to_bf16_kernel<<<(n + 255) / 256, 256, 0, stream>>>(x, xb, n);
  }
  {
    size_t n = (size_t)kDModel * (2 * kDInner);
    transpose_to_bf16_kernel<<<(n + 255) / 256, 256, 0, stream>>>(in_w, wInT, kDModel, 2 * kDInner);
  }
  {
    size_t n = (size_t)kDInner * kDInner;
    transpose_to_bf16_kernel<<<(n + 255) / 256, 256, 0, stream>>>(sdelta_w, wDT, kDInner, kDInner);
  }
  {
    size_t n = (size_t)kDInner * kDModel;
    transpose_to_bf16_kernel<<<(n + 255) / 256, 256, 0, stream>>>(out_w, wOutT, kDInner, kDModel);
  }
  {
    size_t n = (size_t)128 * kDInner;
    pack_bc_kernel<<<(n + 255) / 256, 256, 0, stream>>>(sB_w, sC_w, wBCT);
  }

  // 2) in_proj GEMM: (16384 x 1024) x (1024 x 4096) -> xproj
  {
    dim3 grid((2 * kDInner) / 64, kRows / 256);
    gemm_bf16_kernel<0><<<grid, 256, 0, stream>>>(xb, wInT, xproj, nullptr,
                                                  kRows, 2 * kDInner, kDModel);
  }

  // 3) causal depthwise conv + SiLU -> xc (f32) + xcb (bf16)
  {
    dim3 grid(kDInner / 256, kSeq, kBatch);
    conv_silu_kernel<<<grid, 256, 0, stream>>>(xproj, conv_w, conv_b, xc, xcb);
  }

  // 4) delta = softplus(xc @ sdelta_w + b): (16384 x 2048) x (2048 x 2048)
  {
    dim3 grid(kDInner / 64, kRows / 256);
    gemm_bf16_kernel<1><<<grid, 256, 0, stream>>>(xcb, wDT, delta, sdelta_b,
                                                  kRows, kDInner, kDInner);
  }

  // 5) [B | C] = xc @ [sB_w | sC_w]: (16384 x 2048) x (2048 x 128)
  {
    dim3 grid(128 / 64, kRows / 256);
    gemm_bf16_kernel<0><<<grid, 256, 0, stream>>>(xcb, wBCT, BC, nullptr,
                                                  kRows, 128, kDInner);
  }

  // 6) selective scan + D-skip + gate -> yb (bf16)
  {
    dim3 grid(kBatch, kDInner / 256);
    scan_kernel<<<grid, 256, 0, stream>>>(delta, xc, BC, xproj, A_log, D_param, yb);
  }

  // 7) out = yb @ out_proj_w + out_proj_b: (16384 x 2048) x (2048 x 1024)
  {
    dim3 grid(kDModel / 64, kRows / 256);
    gemm_bf16_kernel<2><<<grid, 256, 0, stream>>>(yb, wOutT, out, out_b,
                                                  kRows, kDModel, kDInner);
  }
}
